// NAFBlock_36919538876991
// MI455X (gfx1250) — matrix operand order repaired and checked by reading, not compiled, not run on hardware
//
#include <hip/hip_runtime.h>
#include <hip/hip_bf16.h>
#include <math.h>

typedef __attribute__((ext_vector_type(16))) __bf16 v16bf;
typedef __attribute__((ext_vector_type(8)))  __bf16 v8bf;
typedef __attribute__((ext_vector_type(2)))  __bf16 v2bf;
typedef __attribute__((ext_vector_type(8)))  float  v8f;

// ---------------- problem constants ----------------
constexpr int NB   = 8;
constexpr int CCH  = 64;
constexpr int HH   = 256;
constexpr int WWI  = 256;
constexpr int HW   = HH * WWI;          // 65536
constexpr float EPS = 1e-6f;

// spatial tiling for pass A/B (needs 3x3 halo)
constexpr int TH   = 4;
constexpr int TWD  = 16;
constexpr int TP   = TH * TWD;          // 64 pixels per tile
constexpr int HPH  = TH + 2;            // 6
constexpr int HPW  = TWD + 2;           // 18
constexpr int HP   = HPH * HPW;         // 108 halo pixels
constexpr int HPP  = 112;               // padded to 7 * 16
constexpr int NPT  = HPP / 16;          // 7 pixel-tiles
constexpr int TILES_W = WWI / TWD;      // 16
constexpr int TILES   = (HH / TH) * TILES_W;  // 1024 tiles per image

// workspace layout (float offsets)
constexpr size_t O_L1M  = 0;
constexpr size_t O_L1R  = 512;
constexpr size_t O_SPOOL= 1024;
constexpr size_t O_SSIG = 1536;
constexpr size_t O_L2M  = 2048;
constexpr size_t O_L2R  = 2560;
constexpr size_t O_SCAP = 3072;                         // 8*1024*64
constexpr size_t O_L2S  = O_SCAP + (size_t)NB*TILES*64; // 527360
constexpr size_t O_L2Q  = O_L2S  + (size_t)NB*TILES*64; // 1051648

// ---------------- helpers (native HW bf16 converts) ----------------
__device__ __forceinline__ __bf16 f2bf(float f) { return (__bf16)f; }
__device__ __forceinline__ float  bf2f(__bf16 h) { return (float)h; }

// A fragment (16x32 bf16): lane holds row M=lane&15.
// e -> K = kofs + half*16 + e
__device__ __forceinline__ v16bf afrag(const float* __restrict__ W, int row, int ld,
                                       int kofs, int half) {
    const float* rp = W + row * ld + kofs;
    v16bf a;
#pragma unroll
    for (int i = 0; i < 8; i++) {
        a[i]     = f2bf(rp[half * 16 + i]);
        a[i + 8] = f2bf(rp[half * 16 + 8 + i]);
    }
    return a;
}

// B fragment (32x16 bf16): lane holds column N (pixel p), elements e -> K = kofs + half*16 + e.
// act layout: [pixel][64 channels], row stride 64 bf16 (128B).
__device__ __forceinline__ v16bf bfrag(const __bf16* __restrict__ act, int p,
                                       int kofs, int half) {
    const __bf16* sp = act + p * 64 + kofs + half * 16;   // 16B aligned
    v8bf lo = *(const v8bf*)sp;
    v8bf hi = *(const v8bf*)(sp + 8);
    return __builtin_shufflevector(lo, hi, 0,1,2,3,4,5,6,7,8,9,10,11,12,13,14,15);
}

#define WMMA_BF16(A, B, C) \
    __builtin_amdgcn_wmma_f32_16x16x32_bf16(false, (A), false, (B), (short)0, (C), false, false)

// ---------------- kernel 1: LN1 stats ----------------
__global__ __launch_bounds__(256) void k_ln1(const float* __restrict__ x,
                                             float* __restrict__ ws) {
    __shared__ float rs[256], rq[256];
    int bid = blockIdx.x;                 // n*64+c
    const float* p = x + (size_t)bid * HW;
    float s = 0.f, q = 0.f;
    for (int i = threadIdx.x; i < HW; i += 256) {
        float v = p[i];
        s += v; q += v * v;
    }
    rs[threadIdx.x] = s; rq[threadIdx.x] = q;
    __syncthreads();
    for (int st = 128; st > 0; st >>= 1) {
        if ((int)threadIdx.x < st) {
            rs[threadIdx.x] += rs[threadIdx.x + st];
            rq[threadIdx.x] += rq[threadIdx.x + st];
        }
        __syncthreads();
    }
    if (threadIdx.x == 0) {
        float m   = rs[0] / (float)HW;
        float var = rq[0] / (float)HW - m * m;
        ws[O_L1M + bid] = m;
        ws[O_L1R + bid] = rsqrtf(var + EPS);
    }
}

// ---------------- kernel 2: pass A (LN1 -> pw1 -> dw3x3 -> gate -> SCA partial sums) ----------------
__global__ __launch_bounds__(256) void k_passA(const float* __restrict__ x,
        const float* __restrict__ ln1w, const float* __restrict__ ln1b,
        const float* __restrict__ w1,   const float* __restrict__ b1,
        const float* __restrict__ dww,  const float* __restrict__ dwb,
        float* __restrict__ ws) {
    __shared__ __align__(16) __bf16 s_act[HPP * 64];    // 14336 B
    __shared__ __align__(16) __bf16 s_z[HPP * 128];     // 28672 B
    __shared__ float s_red0[256], s_red1[256];
    int tid = threadIdx.x;
    int n = blockIdx.x / TILES, t = blockIdx.x % TILES;
    int th0 = (t >> 4) * TH, tw0 = (t & 15) * TWD;

    // stage normalized activations (zero outside image / padding)
    for (int idx = tid; idx < HPP * 64; idx += 256) {
        int p = idx % HPP, c = idx / HPP;
        float xn = 0.f;
        if (p < HP) {
            int hh = th0 - 1 + p / HPW, wp = tw0 - 1 + p % HPW;
            if (hh >= 0 && hh < HH && wp >= 0 && wp < WWI) {
                float v = x[(size_t)(n * CCH + c) * HW + hh * WWI + wp];
                xn = (v - ws[O_L1M + n * CCH + c]) * ws[O_L1R + n * CCH + c] * ln1w[c] + ln1b[c];
            }
        }
        s_act[p * 64 + c] = f2bf(xn);
    }
    __syncthreads();

    // pw1: each wave owns one of the 8 output tiles; A-frags + biases hoisted
    {
        int wave = tid >> 5, lane = tid & 31, lr = lane & 15, half = lane >> 4;
        int ot = wave;
        int ob = ot * 16 + 8 * half;
        v16bf a0 = afrag(w1, ot * 16 + lr, 64, 0,  half);
        v16bf a1 = afrag(w1, ot * 16 + lr, 64, 32, half);
        float bias[8];
#pragma unroll
        for (int r = 0; r < 8; r++) bias[r] = b1[ob + r];
#pragma unroll 1
        for (int pt = 0; pt < NPT; pt++) {
            v16bf bb0 = bfrag(s_act, pt * 16 + lr, 0,  half);
            v16bf bb1 = bfrag(s_act, pt * 16 + lr, 32, half);
            v8f acc = {0.f,0.f,0.f,0.f,0.f,0.f,0.f,0.f};
            acc = WMMA_BF16(a0, bb0, acc);
            acc = WMMA_BF16(a1, bb1, acc);
            int p = pt * 16 + lr;
            int hh = th0 - 1 + p / HPW, wp = tw0 - 1 + p % HPW;
            float m = (p < HP && hh >= 0 && hh < HH && wp >= 0 && wp < WWI) ? 1.f : 0.f;
#pragma unroll
            for (int r = 0; r < 8; r++)
                s_z[p * 128 + ob + r] = f2bf((acc[r] + bias[r]) * m);  // SAME pad: z=0 off-image
        }
    }
    __syncthreads();

    // depthwise 3x3 + simple gate, channel-paired (b32 taps), deterministic pool partial
    {
        int cp = tid >> 3, ps = tid & 7;       // 32 channel pairs x 8 pixel groups
        int c0 = cp * 2;
        float wu0[9], wu1[9], wv0[9], wv1[9];
#pragma unroll
        for (int k = 0; k < 9; k++) {
            wu0[k] = dww[c0 * 9 + k];        wu1[k] = dww[(c0 + 1) * 9 + k];
            wv0[k] = dww[(c0 + 64) * 9 + k]; wv1[k] = dww[(c0 + 65) * 9 + k];
        }
        float bu0 = dwb[c0], bu1 = dwb[c0 + 1], bv0 = dwb[c0 + 64], bv1 = dwb[c0 + 65];
        float a0 = 0.f, a1 = 0.f;
        for (int k = 0; k < 8; k++) {
            int pl = ps * 8 + k;
            int thh = pl >> 4, tww = pl & 15;
            float u0 = bu0, u1 = bu1, v0 = bv0, v1 = bv1;
#pragma unroll
            for (int d = 0; d < 9; d++) {
                int ph = (thh + d / 3) * HPW + (tww + d % 3);
                v2bf pu = *(const v2bf*)&s_z[ph * 128 + c0];
                v2bf pv = *(const v2bf*)&s_z[ph * 128 + c0 + 64];
                u0 += wu0[d] * bf2f(pu[0]); u1 += wu1[d] * bf2f(pu[1]);
                v0 += wv0[d] * bf2f(pv[0]); v1 += wv1[d] * bf2f(pv[1]);
            }
            a0 += u0 * v0;
            a1 += u1 * v1;
        }
        s_red0[tid] = a0;
        s_red1[tid] = a1;
    }
    __syncthreads();
    if (tid < 64) {
        int cp = tid >> 1, par = tid & 1;
        const float* arr = par ? s_red1 : s_red0;
        float s = 0.f;
#pragma unroll
        for (int j = 0; j < 8; j++) s += arr[cp * 8 + j];
        ws[O_SCAP + ((size_t)n * TILES + t) * 64 + tid] = s;
    }
}

// ---------------- kernel 3/4: SCA pool + sigmoid ----------------
__global__ __launch_bounds__(256) void k_sca_pool(float* __restrict__ ws) {
    int id = blockIdx.x * 256 + threadIdx.x;
    if (id >= NB * 64) return;
    int n = id >> 6, c = id & 63;
    float s = 0.f;
    for (int t = 0; t < TILES; t++)
        s += ws[O_SCAP + ((size_t)n * TILES + t) * 64 + c];
    ws[O_SPOOL + id] = s / (float)HW;
}

__global__ __launch_bounds__(256) void k_sca_sig(const float* __restrict__ scaw,
                                                 const float* __restrict__ scab,
                                                 float* __restrict__ ws) {
    int id = blockIdx.x * 256 + threadIdx.x;
    if (id >= NB * 64) return;
    int n = id >> 6, o = id & 63;
    float acc = scab[o];
    for (int c = 0; c < 64; c++)
        acc += scaw[o * 64 + c] * ws[O_SPOOL + n * 64 + c];
    ws[O_SSIG + id] = 1.f / (1.f + expf(-acc));
}

// ---------------- kernel 5: pass B (recompute y, SCA scale, pw2, residual, LN2 partials) ----------------
__global__ __launch_bounds__(256) void k_passB(const float* __restrict__ x,
        const float* __restrict__ ln1w, const float* __restrict__ ln1b,
        const float* __restrict__ w1,   const float* __restrict__ b1,
        const float* __restrict__ dww,  const float* __restrict__ dwb,
        const float* __restrict__ w2,   const float* __restrict__ b2,
        const float* __restrict__ beta1, float* __restrict__ out,
        float* __restrict__ ws) {
    __shared__ __align__(16) __bf16 s_z[HPP * 128];     // 28672 B
    __shared__ __align__(16) float  s_xraw[TP * 64];    // 16384 B (later holds x')
    __shared__ __align__(16) __bf16 s_act[HPP * 64];    // 14336 B (later holds ys)
    __shared__ float s_rs[256], s_rq[256];
    int tid = threadIdx.x;
    int n = blockIdx.x / TILES, t = blockIdx.x % TILES;
    int th0 = (t >> 4) * TH, tw0 = (t & 15) * TWD;
    int wave = tid >> 5, lane = tid & 31, lr = lane & 15, half = lane >> 4;

    for (int idx = tid; idx < HPP * 64; idx += 256) {
        int p = idx % HPP, c = idx / HPP;
        float xn = 0.f;
        if (p < HP) {
            int hh = th0 - 1 + p / HPW, wp = tw0 - 1 + p % HPW;
            if (hh >= 0 && hh < HH && wp >= 0 && wp < WWI) {
                float v = x[(size_t)(n * CCH + c) * HW + hh * WWI + wp];
                xn = (v - ws[O_L1M + n * CCH + c]) * ws[O_L1R + n * CCH + c] * ln1w[c] + ln1b[c];
                int lh = hh - th0, lw = wp - tw0;
                if (lh >= 0 && lh < TH && lw >= 0 && lw < TWD)
                    s_xraw[(lh * TWD + lw) * 64 + c] = v;
            }
        }
        s_act[p * 64 + c] = f2bf(xn);
    }
    __syncthreads();

    // pw1 (same as pass A): wave = output tile, A-frags hoisted
    {
        int ot = wave;
        int ob = ot * 16 + 8 * half;
        v16bf a0 = afrag(w1, ot * 16 + lr, 64, 0,  half);
        v16bf a1 = afrag(w1, ot * 16 + lr, 64, 32, half);
        float bias[8];
#pragma unroll
        for (int r = 0; r < 8; r++) bias[r] = b1[ob + r];
#pragma unroll 1
        for (int pt = 0; pt < NPT; pt++) {
            v16bf bb0 = bfrag(s_act, pt * 16 + lr, 0,  half);
            v16bf bb1 = bfrag(s_act, pt * 16 + lr, 32, half);
            v8f acc = {0.f,0.f,0.f,0.f,0.f,0.f,0.f,0.f};
            acc = WMMA_BF16(a0, bb0, acc);
            acc = WMMA_BF16(a1, bb1, acc);
            int p = pt * 16 + lr;
            int hh = th0 - 1 + p / HPW, wp = tw0 - 1 + p % HPW;
            float m = (p < HP && hh >= 0 && hh < HH && wp >= 0 && wp < WWI) ? 1.f : 0.f;
#pragma unroll
            for (int r = 0; r < 8; r++)
                s_z[p * 128 + ob + r] = f2bf((acc[r] + bias[r]) * m);
        }
    }
    __syncthreads();

    // dw + gate + SCA scale -> ys (reuse s_act), channel-paired
    {
        int cp = tid >> 3, ps = tid & 7;
        int c0 = cp * 2;
        float wu0[9], wu1[9], wv0[9], wv1[9];
#pragma unroll
        for (int k = 0; k < 9; k++) {
            wu0[k] = dww[c0 * 9 + k];        wu1[k] = dww[(c0 + 1) * 9 + k];
            wv0[k] = dww[(c0 + 64) * 9 + k]; wv1[k] = dww[(c0 + 65) * 9 + k];
        }
        float bu0 = dwb[c0], bu1 = dwb[c0 + 1], bv0 = dwb[c0 + 64], bv1 = dwb[c0 + 65];
        float sw0 = ws[O_SSIG + n * 64 + c0], sw1 = ws[O_SSIG + n * 64 + c0 + 1];
        for (int k = 0; k < 8; k++) {
            int pl = ps * 8 + k;
            int thh = pl >> 4, tww = pl & 15;
            float u0 = bu0, u1 = bu1, v0 = bv0, v1 = bv1;
#pragma unroll
            for (int d = 0; d < 9; d++) {
                int ph = (thh + d / 3) * HPW + (tww + d % 3);
                v2bf pu = *(const v2bf*)&s_z[ph * 128 + c0];
                v2bf pv = *(const v2bf*)&s_z[ph * 128 + c0 + 64];
                u0 += wu0[d] * bf2f(pu[0]); u1 += wu1[d] * bf2f(pu[1]);
                v0 += wv0[d] * bf2f(pv[0]); v1 += wv1[d] * bf2f(pv[1]);
            }
            v2bf ys;
            ys[0] = f2bf(u0 * v0 * sw0);
            ys[1] = f2bf(u1 * v1 * sw1);
            *(v2bf*)&s_act[pl * 64 + c0] = ys;
        }
    }
    __syncthreads();

    // pw2 (64x64) + residual; ot fixed per wave (wave&3), 2 pixel-tiles per wave
    {
        int ot = wave & 3;
        int ob = ot * 16 + 8 * half;
        v16bf a0 = afrag(w2, ot * 16 + lr, 64, 0,  half);
        v16bf a1 = afrag(w2, ot * 16 + lr, 64, 32, half);
        float bias[8], bet[8];
#pragma unroll
        for (int r = 0; r < 8; r++) { bias[r] = b2[ob + r]; bet[r] = beta1[ob + r]; }
#pragma unroll 1
        for (int j = 0; j < 2; j++) {
            int pt = (wave >> 2) + 2 * j;
            v16bf bb0 = bfrag(s_act, pt * 16 + lr, 0,  half);
            v16bf bb1 = bfrag(s_act, pt * 16 + lr, 32, half);
            v8f acc = {0.f,0.f,0.f,0.f,0.f,0.f,0.f,0.f};
            acc = WMMA_BF16(a0, bb0, acc);
            acc = WMMA_BF16(a1, bb1, acc);
            int pl = pt * 16 + lr;
            int thh = pl >> 4, tww = pl & 15;
            size_t pix = (size_t)(th0 + thh) * WWI + (tw0 + tww);
#pragma unroll
            for (int r = 0; r < 8; r++) {
                int o = ob + r;
                float val = s_xraw[pl * 64 + o] + bet[r] * (acc[r] + bias[r]);
                out[(size_t)(n * CCH + o) * HW + pix] = val;
                s_xraw[pl * 64 + o] = val;
            }
        }
    }
    __syncthreads();

    // deterministic LN2 partial sums of x'
    {
        int c = tid >> 2, ps = tid & 3;
        float s = 0.f, q = 0.f;
        for (int k = 0; k < 16; k++) {
            float v = s_xraw[(ps * 16 + k) * 64 + c];
            s += v; q += v * v;
        }
        s_rs[tid] = s; s_rq[tid] = q;
    }
    __syncthreads();
    if (tid < 64) {
        float s = s_rs[4*tid] + s_rs[4*tid+1] + s_rs[4*tid+2] + s_rs[4*tid+3];
        float q = s_rq[4*tid] + s_rq[4*tid+1] + s_rq[4*tid+2] + s_rq[4*tid+3];
        size_t o = ((size_t)n * TILES + t) * 64 + tid;
        ws[O_L2S + o] = s;
        ws[O_L2Q + o] = q;
    }
}

// ---------------- kernel 6: LN2 finalize ----------------
__global__ __launch_bounds__(256) void k_ln2fin(float* __restrict__ ws) {
    int id = blockIdx.x * 256 + threadIdx.x;
    if (id >= NB * 64) return;
    int n = id >> 6, c = id & 63;
    float s = 0.f, q = 0.f;
    for (int t = 0; t < TILES; t++) {
        size_t o = ((size_t)n * TILES + t) * 64 + c;
        s += ws[O_L2S + o];
        q += ws[O_L2Q + o];
    }
    float m = s / (float)HW;
    float var = q / (float)HW - m * m;
    ws[O_L2M + id] = m;
    ws[O_L2R + id] = rsqrtf(var + EPS);
}

// ---------------- kernel 7: pass C (LN2 -> pw3 -> gate -> pw4 -> residual, in place) ----------------
__global__ __launch_bounds__(256) void k_passC(float* __restrict__ out,
        const float* __restrict__ ln2w, const float* __restrict__ ln2b,
        const float* __restrict__ w3,   const float* __restrict__ b3,
        const float* __restrict__ w4,   const float* __restrict__ b4,
        const float* __restrict__ beta2, const float* __restrict__ ws) {
    __shared__ __align__(16) __bf16 s_act[128 * 64];    // 16384 B (act, then gated t)
    __shared__ __align__(16) __bf16 s_z3[128 * 128];    // 32768 B
    int tid = threadIdx.x;
    int n = blockIdx.x >> 9;
    int p0 = (blockIdx.x & 511) * 128;
    int wave = tid >> 5, lane = tid & 31, lr = lane & 15, half = lane >> 4;

    for (int idx = tid; idx < 128 * 64; idx += 256) {
        int p = idx & 127, c = idx >> 7;
        float v = out[(size_t)(n * CCH + c) * HW + p0 + p];
        float xn = (v - ws[O_L2M + n * 64 + c]) * ws[O_L2R + n * 64 + c] * ln2w[c] + ln2b[c];
        s_act[p * 64 + c] = f2bf(xn);
    }
    __syncthreads();

    // pw3: 128 outs; wave = output tile, 8 pixel-tiles each
    {
        int ot = wave;
        int ob = ot * 16 + 8 * half;
        v16bf a0 = afrag(w3, ot * 16 + lr, 64, 0,  half);
        v16bf a1 = afrag(w3, ot * 16 + lr, 64, 32, half);
        float bias[8];
#pragma unroll
        for (int r = 0; r < 8; r++) bias[r] = b3[ob + r];
#pragma unroll 1
        for (int pt = 0; pt < 8; pt++) {
            v16bf bb0 = bfrag(s_act, pt * 16 + lr, 0,  half);
            v16bf bb1 = bfrag(s_act, pt * 16 + lr, 32, half);
            v8f acc = {0.f,0.f,0.f,0.f,0.f,0.f,0.f,0.f};
            acc = WMMA_BF16(a0, bb0, acc);
            acc = WMMA_BF16(a1, bb1, acc);
            int p = pt * 16 + lr;
#pragma unroll
            for (int r = 0; r < 8; r++)
                s_z3[p * 128 + ob + r] = f2bf(acc[r] + bias[r]);
        }
    }
    __syncthreads();

    // simple gate -> t (reuse s_act), channel-paired b32 ops
    for (int idx = tid; idx < 128 * 32; idx += 256) {
        int p = idx & 127, cp = idx >> 7;
        int c0 = cp * 2;
        v2bf zu = *(const v2bf*)&s_z3[p * 128 + c0];
        v2bf zv = *(const v2bf*)&s_z3[p * 128 + 64 + c0];
        v2bf tv;
        tv[0] = f2bf(bf2f(zu[0]) * bf2f(zv[0]));
        tv[1] = f2bf(bf2f(zu[1]) * bf2f(zv[1]));
        *(v2bf*)&s_act[p * 64 + c0] = tv;
    }
    __syncthreads();

    // pw4 (64x64) + final residual, in place on out
    {
        int ot = wave & 3;
        int ob = ot * 16 + 8 * half;
        v16bf a0 = afrag(w4, ot * 16 + lr, 64, 0,  half);
        v16bf a1 = afrag(w4, ot * 16 + lr, 64, 32, half);
        float bias[8], bet[8];
#pragma unroll
        for (int r = 0; r < 8; r++) { bias[r] = b4[ob + r]; bet[r] = beta2[ob + r]; }
#pragma unroll 1
        for (int j = 0; j < 4; j++) {
            int pt = (wave >> 2) + 2 * j;
            v16bf bb0 = bfrag(s_act, pt * 16 + lr, 0,  half);
            v16bf bb1 = bfrag(s_act, pt * 16 + lr, 32, half);
            v8f acc = {0.f,0.f,0.f,0.f,0.f,0.f,0.f,0.f};
            acc = WMMA_BF16(a0, bb0, acc);
            acc = WMMA_BF16(a1, bb1, acc);
            int p = pt * 16 + lr;
#pragma unroll
            for (int r = 0; r < 8; r++) {
                int o = ob + r;
                size_t gi = (size_t)(n * CCH + o) * HW + p0 + p;
                out[gi] = out[gi] + bet[r] * (acc[r] + bias[r]);
            }
        }
    }
}

// ---------------- launch ----------------
extern "C" void kernel_launch(void* const* d_in, const int* in_sizes, int n_in,
                              void* d_out, int out_size, void* d_ws, size_t ws_size,
                              hipStream_t stream) {
    const float* x     = (const float*)d_in[0];
    const float* ln1w  = (const float*)d_in[1];
    const float* ln1b  = (const float*)d_in[2];
    const float* w1    = (const float*)d_in[3];
    const float* b1    = (const float*)d_in[4];
    const float* dww   = (const float*)d_in[5];
    const float* dwb   = (const float*)d_in[6];
    const float* scaw  = (const float*)d_in[7];
    const float* scab  = (const float*)d_in[8];
    const float* w2    = (const float*)d_in[9];
    const float* b2    = (const float*)d_in[10];
    const float* ln2w  = (const float*)d_in[11];
    const float* ln2b  = (const float*)d_in[12];
    const float* w3    = (const float*)d_in[13];
    const float* b3    = (const float*)d_in[14];
    const float* w4    = (const float*)d_in[15];
    const float* b4    = (const float*)d_in[16];
    const float* beta1 = (const float*)d_in[17];
    const float* beta2 = (const float*)d_in[18];
    float* out = (float*)d_out;
    float* ws  = (float*)d_ws;

    k_ln1   <<<NB * CCH,    256, 0, stream>>>(x, ws);
    k_passA <<<NB * TILES,  256, 0, stream>>>(x, ln1w, ln1b, w1, b1, dww, dwb, ws);
    k_sca_pool<<<2, 256, 0, stream>>>(ws);
    k_sca_sig <<<2, 256, 0, stream>>>(scaw, scab, ws);
    k_passB <<<NB * TILES,  256, 0, stream>>>(x, ln1w, ln1b, w1, b1, dww, dwb,
                                              w2, b2, beta1, out, ws);
    k_ln2fin<<<2, 256, 0, stream>>>(ws);
    k_passC <<<NB * HW / 128, 256, 0, stream>>>(out, ln2w, ln2b, w3, b3, w4, b4, beta2, ws);
}
